// Interaction_16449724744296
// MI455X (gfx1250) — compile-verified, matrix operand order checked by reading, not run on hardware
//
#include <hip/hip_runtime.h>

// SchNet interaction block for MI455X (gfx1250, wave32).
// All GEMMs via v_wmma_f32_16x16x32_f16 (f16 in, f32 accumulate).
// Edge MLP + gather + scatter-atomic fused: no h/msg traffic to HBM.
// Roofline: ~470 MB unavoidable traffic @ 23.3 TB/s ~= 20 us; WMMA keeps the
// 22 GFLOP of GEMMs well under that, so the kernel stays memory/atomic-bound.
// Hot kernels handle only full 16-row tiles (unguarded); scalar tail kernels
// are launched only when N or E is not a multiple of 16 (never, here).

typedef __attribute__((ext_vector_type(16))) _Float16 v16h;
typedef __attribute__((ext_vector_type(8)))  _Float16 v8h;
typedef __attribute__((ext_vector_type(8)))  float    v8f;
typedef __attribute__((ext_vector_type(4)))  float    v4f;

#define STG_LD 72  // f16 staging row stride (pad; 144B rows stay 16B aligned)

static __device__ inline float ssp(float x) {
    float hx = 0.5f * x;
    return (hx > 14.0f) ? x : 2.0f * __logf(1.0f + __expf(hx));
}

static __device__ inline v8f wmma_f16(v16h a, v16h b, v8f c) {
    return __builtin_amdgcn_wmma_f32_16x16x32_f16(
        false, a, false, b, (short)0, c, false, false);
}

// A-fragment (16x32 f16) from global f32 row-major; p pre-offset to (m, kb):
// entries 0..7 = K kb..kb+7, entries 8..15 = K kb+16..kb+23 (ISA layout).
static __device__ inline v16h load_a_g(const float* __restrict__ p) {
    v4f x0 = *(const v4f*)(p);
    v4f x1 = *(const v4f*)(p + 4);
    v4f x2 = *(const v4f*)(p + 16);
    v4f x3 = *(const v4f*)(p + 20);
    v16h a;
#pragma unroll
    for (int i = 0; i < 4; ++i) {
        a[i]      = (_Float16)x0[i];
        a[4 + i]  = (_Float16)x1[i];
        a[8 + i]  = (_Float16)x2[i];
        a[12 + i] = (_Float16)x3[i];
    }
    return a;
}

// A/B fragment halves from LDS f16 (16B-aligned): p[0..7] and p[16..23].
static __device__ inline v16h load_frag_lds(const _Float16* p) {
    v8h x0 = *(const v8h*)(p);
    v8h x1 = *(const v8h*)(p + 16);
    v16h r;
#pragma unroll
    for (int i = 0; i < 8; ++i) { r[i] = x0[i]; r[8 + i] = x1[i]; }
    return r;
}

// ---------------------------------------------------------------- kernel 0
__global__ __launch_bounds__(256) void k_zero4(v4f* __restrict__ p, int n4) {
    unsigned i = blockIdx.x * 256u + threadIdx.x;
    if (i < (unsigned)n4) p[i] = (v4f){0.f, 0.f, 0.f, 0.f};
}

// ------------------------------------------------- kernel 1: node @ W1^T
__global__ __launch_bounds__(256) void k_node_lin(
    const float* __restrict__ node, const float* __restrict__ W1,
    float* __restrict__ nn, int fullTiles)
{
    __shared__ _Float16 sW[64 * 64];
    int tid = threadIdx.x;
    for (int i = tid; i < 64 * 64; i += 256) sW[i] = (_Float16)W1[i];
    __syncthreads();

    int wave = tid >> 5, lane = tid & 31;
    int m = lane & 15, hi = lane >> 4, kb = hi * 8, nsel = lane & 15;
    int step = gridDim.x * 8;

    for (int tile = blockIdx.x * 8 + wave; tile < fullTiles; tile += step) {
        int row0 = tile * 16;
        const float* p = node + (unsigned)((row0 + m) * 64 + kb);
        v16h a[2];
#pragma unroll
        for (int k = 0; k < 2; ++k) a[k] = load_a_g(p + k * 32);

#pragma unroll
        for (int n = 0; n < 4; ++n) {
            v8f acc = {0.f, 0.f, 0.f, 0.f, 0.f, 0.f, 0.f, 0.f};
#pragma unroll
            for (int k = 0; k < 2; ++k) {
                v16h b = load_frag_lds(sW + (n * 16 + nsel) * 64 + k * 32 + kb);
                acc = wmma_f16(a[k], b, acc);
            }
            unsigned base = (unsigned)((row0 + 8 * hi) * 64 + n * 16 + nsel);
#pragma unroll
            for (int r = 0; r < 8; ++r) nn[base + (unsigned)(r * 64)] = acc[r];
        }
    }
}

// --------- kernel 2: fused edge MLP + gather(new_node[src]) * h + scatter-add
__global__ __launch_bounds__(256) void k_edge(
    const float* __restrict__ rbf,
    const int* __restrict__ src, const int* __restrict__ dst,
    const float* __restrict__ Wc1, const float* __restrict__ bc1,
    const float* __restrict__ Wc2, const float* __restrict__ bc2,
    const float* __restrict__ nn, float* __restrict__ cf, int fullTiles)
{
    __shared__ _Float16 sWc1[64 * 128];
    __shared__ _Float16 sWc2[64 * 64];
    __shared__ _Float16 stage[8][16 * STG_LD];

    int tid = threadIdx.x;
    for (int i = tid; i < 64 * 128; i += 256) sWc1[i] = (_Float16)Wc1[i];
    for (int i = tid; i < 64 * 64;  i += 256) sWc2[i] = (_Float16)Wc2[i];
    __syncthreads();

    int wave = tid >> 5, lane = tid & 31;
    int m = lane & 15, hi = lane >> 4, kb = hi * 8, nsel = lane & 15;
    _Float16* st = stage[wave];

    float bv1[4], bv2[4];
#pragma unroll
    for (int n = 0; n < 4; ++n) { bv1[n] = bc1[n * 16 + nsel]; bv2[n] = bc2[n * 16 + nsel]; }

    int step = gridDim.x * 8;

    for (int tile = blockIdx.x * 8 + wave; tile < fullTiles; tile += step) {
        int e0 = tile * 16;

        // A1: 16x128 f16 of rbf rows (4 K-steps)
        const float* rp = rbf + (unsigned)((e0 + m) * 128 + kb);
        v16h a1[4];
#pragma unroll
        for (int k = 0; k < 4; ++k) a1[k] = load_a_g(rp + k * 32);

        // edge indices for the 8 rows this lane owns (C-layout rows)
        unsigned soff[8], doff[8];
#pragma unroll
        for (int r = 0; r < 8; ++r) {
            int e = e0 + r + 8 * hi;
            soff[r] = (unsigned)src[e] * 64u;
            doff[r] = (unsigned)dst[e] * 64u;
        }

        // GEMM1 + bias + ssp -> f16 staging (C-layout -> row-major restripe)
#pragma unroll
        for (int n = 0; n < 4; ++n) {
            float bv = bv1[n];
            v8f acc = {bv, bv, bv, bv, bv, bv, bv, bv};
#pragma unroll
            for (int k = 0; k < 4; ++k) {
                v16h b = load_frag_lds(sWc1 + (n * 16 + nsel) * 128 + k * 32 + kb);
                acc = wmma_f16(a1[k], b, acc);
            }
#pragma unroll
            for (int r = 0; r < 8; ++r)
                st[(r + 8 * hi) * STG_LD + n * 16 + nsel] = (_Float16)ssp(acc[r]);
        }
        // same-wave LDS RAW -> compiler inserts s_wait_dscnt; no barrier needed

        // A2: 16x64 from staging (2 K-steps)
        v16h a2[2];
#pragma unroll
        for (int k = 0; k < 2; ++k)
            a2[k] = load_frag_lds(st + m * STG_LD + k * 32 + kb);

        // GEMM2 + bias, then fused u_mul_e + segment_sum (f32 atomics -> L2)
#pragma unroll
        for (int n = 0; n < 4; ++n) {
            float bv = bv2[n];
            v8f acc = {bv, bv, bv, bv, bv, bv, bv, bv};
#pragma unroll
            for (int k = 0; k < 2; ++k) {
                v16h b = load_frag_lds(sWc2 + (n * 16 + nsel) * 64 + k * 32 + kb);
                acc = wmma_f16(a2[k], b, acc);
            }
            unsigned col = (unsigned)(n * 16 + nsel);
#pragma unroll
            for (int r = 0; r < 8; ++r) {
                float msg = nn[soff[r] + col] * acc[r];
                atomicAdd(cf + (doff[r] + col), msg);
            }
        }
    }
}

// ------- kernel 3: out = node + ssp(cf@W2^T + b2)@W3^T + b3  (in place on cf)
__global__ __launch_bounds__(256) void k_final(
    const float* __restrict__ node,
    const float* __restrict__ W2, const float* __restrict__ b2,
    const float* __restrict__ W3, const float* __restrict__ b3,
    float* __restrict__ out /* holds cf on entry */, int fullTiles)
{
    __shared__ _Float16 sW2[64 * 64];
    __shared__ _Float16 sW3[64 * 64];
    __shared__ _Float16 stage[8][16 * STG_LD];

    int tid = threadIdx.x;
    for (int i = tid; i < 64 * 64; i += 256) { sW2[i] = (_Float16)W2[i]; sW3[i] = (_Float16)W3[i]; }
    __syncthreads();

    int wave = tid >> 5, lane = tid & 31;
    int m = lane & 15, hi = lane >> 4, kb = hi * 8, nsel = lane & 15;
    _Float16* st = stage[wave];

    float bv2[4], bv3[4];
#pragma unroll
    for (int n = 0; n < 4; ++n) { bv2[n] = b2[n * 16 + nsel]; bv3[n] = b3[n * 16 + nsel]; }

    int step = gridDim.x * 8;

    for (int tile = blockIdx.x * 8 + wave; tile < fullTiles; tile += step) {
        int row0 = tile * 16;
        const float* p = out + (unsigned)((row0 + m) * 64 + kb);  // cf row
        v16h a[2];
#pragma unroll
        for (int k = 0; k < 2; ++k) a[k] = load_a_g(p + k * 32);

#pragma unroll
        for (int n = 0; n < 4; ++n) {
            float bv = bv2[n];
            v8f acc = {bv, bv, bv, bv, bv, bv, bv, bv};
#pragma unroll
            for (int k = 0; k < 2; ++k) {
                v16h b = load_frag_lds(sW2 + (n * 16 + nsel) * 64 + k * 32 + kb);
                acc = wmma_f16(a[k], b, acc);
            }
#pragma unroll
            for (int r = 0; r < 8; ++r)
                st[(r + 8 * hi) * STG_LD + n * 16 + nsel] = (_Float16)ssp(acc[r]);
        }

        v16h a2[2];
#pragma unroll
        for (int k = 0; k < 2; ++k)
            a2[k] = load_frag_lds(st + m * STG_LD + k * 32 + kb);

#pragma unroll
        for (int n = 0; n < 4; ++n) {
            float bv = bv3[n];
            v8f acc = {bv, bv, bv, bv, bv, bv, bv, bv};
#pragma unroll
            for (int k = 0; k < 2; ++k) {
                v16h b = load_frag_lds(sW3 + (n * 16 + nsel) * 64 + k * 32 + kb);
                acc = wmma_f16(a2[k], b, acc);
            }
            unsigned base = (unsigned)((row0 + 8 * hi) * 64 + n * 16 + nsel);
#pragma unroll
            for (int r = 0; r < 8; ++r) {
                unsigned off = base + (unsigned)(r * 64);
                out[off] = node[off] + acc[r];
            }
        }
    }
}

// ---------------- scalar tail kernels (only launched when count % 16 != 0) --
__global__ __launch_bounds__(64) void k_node_tail(
    const float* __restrict__ node, const float* __restrict__ W1,
    float* __restrict__ nn, int start)
{
    int row = start + blockIdx.x, col = threadIdx.x;
    float s = 0.f;
    for (int k = 0; k < 64; ++k) s += node[row * 64 + k] * W1[col * 64 + k];
    nn[row * 64 + col] = s;
}

__global__ __launch_bounds__(64) void k_edge_tail(
    const float* __restrict__ rbf,
    const int* __restrict__ src, const int* __restrict__ dst,
    const float* __restrict__ Wc1, const float* __restrict__ bc1,
    const float* __restrict__ Wc2, const float* __restrict__ bc2,
    const float* __restrict__ nn, float* __restrict__ cf, int start)
{
    __shared__ float t[64];
    int e = start + blockIdx.x, j = threadIdx.x;
    float s = bc1[j];
    for (int k = 0; k < 128; ++k) s += rbf[e * 128 + k] * Wc1[j * 128 + k];
    t[j] = ssp(s);
    __syncthreads();
    float h = bc2[j];
    for (int k = 0; k < 64; ++k) h += t[k] * Wc2[j * 64 + k];
    float msg = nn[src[e] * 64 + j] * h;
    atomicAdd(cf + (dst[e] * 64 + j), msg);
}

__global__ __launch_bounds__(64) void k_final_tail(
    const float* __restrict__ node,
    const float* __restrict__ W2, const float* __restrict__ b2,
    const float* __restrict__ W3, const float* __restrict__ b3,
    float* __restrict__ out, int start)
{
    __shared__ float t[64];
    int row = start + blockIdx.x, j = threadIdx.x;
    float s = b2[j];
    for (int k = 0; k < 64; ++k) s += out[row * 64 + k] * W2[j * 64 + k];
    t[j] = ssp(s);
    __syncthreads();
    float o = b3[j];
    for (int k = 0; k < 64; ++k) o += t[k] * W3[j * 64 + k];
    out[row * 64 + j] = node[row * 64 + j] + o;
}

extern "C" void kernel_launch(void* const* d_in, const int* in_sizes, int n_in,
                              void* d_out, int out_size, void* d_ws, size_t ws_size,
                              hipStream_t stream) {
    const float* node = (const float*)d_in[0];
    const float* rbf  = (const float*)d_in[1];
    const int*   src  = (const int*)d_in[2];
    const int*   dst  = (const int*)d_in[3];
    const float* W1   = (const float*)d_in[4];
    const float* Wc1  = (const float*)d_in[5];
    const float* bc1  = (const float*)d_in[6];
    const float* Wc2  = (const float*)d_in[7];
    const float* bc2  = (const float*)d_in[8];
    const float* W2   = (const float*)d_in[9];
    const float* b2   = (const float*)d_in[10];
    const float* W3   = (const float*)d_in[11];
    const float* b3   = (const float*)d_in[12];

    int N = in_sizes[0] / 64;    // 100000
    int E = in_sizes[2];         // 800000
    float* out = (float*)d_out;  // doubles as cf accumulator
    float* nn  = (float*)d_ws;   // new_node scratch: N*64 floats

    int n4 = N * 16;             // N*64 floats / 4
    k_zero4<<<(n4 + 255) / 256, 256, 0, stream>>>((v4f*)out, n4);

    int nTiles = N / 16, nRem = N % 16;   // 6250, 0
    int eTiles = E / 16, eRem = E % 16;   // 50000, 0
    // ~2 tiles/wave for node kernels, ~4 for the edge kernel: amortizes the
    // per-block weight->f16 LDS conversion while keeping >10k waves in flight.
    int nBlocks = (nTiles + 15) / 16;
    int eBlocks = (eTiles + 31) / 32;

    if (nTiles) k_node_lin<<<nBlocks, 256, 0, stream>>>(node, W1, nn, nTiles);
    if (nRem)   k_node_tail<<<nRem, 64, 0, stream>>>(node, W1, nn, nTiles * 16);

    if (eTiles) k_edge<<<eBlocks, 256, 0, stream>>>(rbf, src, dst, Wc1, bc1,
                                                    Wc2, bc2, nn, out, eTiles);
    if (eRem)   k_edge_tail<<<eRem, 64, 0, stream>>>(rbf, src, dst, Wc1, bc1,
                                                     Wc2, bc2, nn, out, eTiles * 16);

    if (nTiles) k_final<<<nBlocks, 256, 0, stream>>>(node, W2, b2, W3, b3, out, nTiles);
    if (nRem)   k_final_tail<<<nRem, 64, 0, stream>>>(node, W2, b2, W3, b3, out, nTiles * 16);
}
